// RoIHeads_62723702391391
// MI455X (gfx1250) — compile-verified, matrix operand order checked by reading, not run on hardware
//
#include <hip/hip_runtime.h>
#include <hip/hip_bf16.h>
#include <math.h>

// ---------------------------------------------------------------------------
// Faster-RCNN RoIHeads postprocess for MI455X (gfx1250, wave32).
//   B=4 images, N=2000 rois, C=81 classes, top-4096 pre-NMS, 100 dets/img.
// Pipeline: softmax+decode -> histogram select -> compact -> bitonic top-k
//           -> WMMA-assisted NMS bitmask -> single-wave greedy scan -> emit.
// ---------------------------------------------------------------------------

#define BIMG 4
#define NROI 2000
#define NCLS 81
#define NCAND (NROI * (NCLS - 1))   // 160000 per image
#define KTOP 4096                   // PRE_NMS_K
#define MSORT 8192                  // sort buffer (power of 2)
#define NDET 100
#define NBINS 4096
#define SCORE_THRESH 0.05f
#define MIN_SIZE 0.01f
#define BBOX_CLIP 4.135166556742356f  // log(1000/16)

typedef float v2f  __attribute__((ext_vector_type(2)));
typedef float v8f  __attribute__((ext_vector_type(8)));
typedef _Float16 v16h __attribute__((ext_vector_type(16)));

#if defined(__has_builtin)
#if __has_builtin(__builtin_amdgcn_wmma_f32_16x16x4_f32)
#define WMMA_MODE 2   // native f32 16x16x4 outer-sum
#elif __has_builtin(__builtin_amdgcn_wmma_f32_16x16x32_f16)
#define WMMA_MODE 1   // probe-confirmed f16 path (areas scaled by 1/64)
#endif
#endif
#ifndef WMMA_MODE
#define WMMA_MODE 0
#endif

// ---------------------------------------------------------------------------
// K1: per (b,n) row — softmax over 81 classes, decode 80 class boxes, clip,
// validity mask. Writes masked scores [B][160000] and boxes [B][160000][4].
// ---------------------------------------------------------------------------
__global__ __launch_bounds__(128) void k_score_decode(
    const float* __restrict__ logits, const float* __restrict__ regs,
    const float* __restrict__ rois, const int* __restrict__ pH,
    const int* __restrict__ pW, float* __restrict__ scores,
    float4* __restrict__ boxes) {
  const int row = blockIdx.x;              // b*NROI + n
  const int b = row / NROI, n = row % NROI;
  const int t = threadIdx.x;
  __shared__ float red[128];

  if (t == 0) __builtin_prefetch(&regs[(size_t)row * NCLS * 4], 0, 0);

  float l = -1e30f;
  if (t < NCLS) l = logits[row * NCLS + t];
  red[t] = l;
  __syncthreads();
  for (int s = 64; s > 0; s >>= 1) {
    if (t < s) red[t] = fmaxf(red[t], red[t + s]);
    __syncthreads();
  }
  const float mx = red[0];
  __syncthreads();
  const float ex = (t < NCLS) ? expf(l - mx) : 0.0f;
  red[t] = ex;
  __syncthreads();
  for (int s = 64; s > 0; s >>= 1) {
    if (t < s) red[t] += red[t + s];
    __syncthreads();
  }
  const float inv = 1.0f / red[0];

  if (t >= 1 && t < NCLS) {               // class c = t (skip background 0)
    const float p = ex * inv;
    const float x1 = rois[row * 4 + 0], y1 = rois[row * 4 + 1];
    const float x2 = rois[row * 4 + 2], y2 = rois[row * 4 + 3];
    const float w = x2 - x1, h = y2 - y1;
    const float cx = x1 + 0.5f * w, cy = y1 + 0.5f * h;
    const float4 d = ((const float4*)regs)[row * NCLS + t];
    const float dw = fminf(d.z, BBOX_CLIP), dh = fminf(d.w, BBOX_CLIP);
    const float pcx = d.x * w + cx, pcy = d.y * h + cy;
    const float pw = expf(dw) * w, ph = expf(dh) * h;
    const float Wf = (float)(*pW), Hf = (float)(*pH);
    const float ox1 = fminf(fmaxf(pcx - 0.5f * pw, 0.f), Wf);
    const float oy1 = fminf(fmaxf(pcy - 0.5f * ph, 0.f), Hf);
    const float ox2 = fminf(fmaxf(pcx + 0.5f * pw, 0.f), Wf);
    const float oy2 = fminf(fmaxf(pcy + 0.5f * ph, 0.f), Hf);
    const bool valid = (p > SCORE_THRESH) && ((ox2 - ox1) >= MIN_SIZE) &&
                       ((oy2 - oy1) >= MIN_SIZE);
    const int e = n * (NCLS - 1) + (t - 1);
    scores[b * NCAND + e] = valid ? p : -1.0f;
    boxes[(size_t)b * NCAND + e] = make_float4(ox1, oy1, ox2, oy2);
  }
}

// ---------------------------------------------------------------------------
// K2: per-image score histogram (4096 bins over (0,1]) -> threshold bin T so
// that #scores in bins >= T covers the top-4096 exactly. Also inits the
// candidate key buffer to sentinel and zeroes the compaction counter.
// ---------------------------------------------------------------------------
__global__ __launch_bounds__(1024) void k_hist_select(
    const float* __restrict__ scores, int* __restrict__ selBin,
    unsigned* __restrict__ cand_count,
    unsigned long long* __restrict__ cand_keys) {
  const int b = blockIdx.x, t = threadIdx.x;
  __shared__ unsigned hist[NBINS];
  for (int i = t; i < NBINS; i += 1024) hist[i] = 0u;
  __syncthreads();
  for (int e = t; e < NCAND; e += 1024) {
    const float s = scores[b * NCAND + e];
    if (s > SCORE_THRESH) {
      int bin = (int)(s * (float)NBINS);
      bin = bin > NBINS - 1 ? NBINS - 1 : bin;
      atomicAdd(&hist[bin], 1u);
    }
  }
  __syncthreads();
  if (t == 0) {
    unsigned acc = 0; int T = 0;
    for (int bin = NBINS - 1; bin >= 0; --bin) {
      acc += hist[bin];
      if (acc >= KTOP) { T = bin; break; }
    }
    selBin[b] = T;
    cand_count[b] = 0u;
  }
  for (int p = t; p < MSORT; p += 1024)
    cand_keys[b * MSORT + p] = ~0ULL;  // sentinel sorts last
}

// ---------------------------------------------------------------------------
// K3: compact candidates (score>thresh, bin>=T) as packed sort keys:
//   key = (~orderable(score) << 32) | index  -> ascending == score desc, idx asc
// ---------------------------------------------------------------------------
__global__ __launch_bounds__(256) void k_compact(
    const float* __restrict__ scores, const int* __restrict__ selBin,
    unsigned* __restrict__ cand_count,
    unsigned long long* __restrict__ cand_keys) {
  const int g = blockIdx.x * 256 + threadIdx.x;
  if (g >= BIMG * NCAND) return;
  const int b = g / NCAND;
  const float s = scores[g];
  if (s > SCORE_THRESH) {
    int bin = (int)(s * (float)NBINS);
    bin = bin > NBINS - 1 ? NBINS - 1 : bin;
    if (bin >= selBin[b]) {
      const unsigned pos = atomicAdd(&cand_count[b], 1u);
      if (pos < MSORT) {
        unsigned u = __float_as_uint(s);
        u = (u & 0x80000000u) ? ~u : (u | 0x80000000u);
        cand_keys[b * MSORT + pos] =
            ((unsigned long long)(~u) << 32) | (unsigned)(g - b * NCAND);
      }
    }
  }
}

// ---------------------------------------------------------------------------
// K4: one block per image — in-LDS bitonic sort of 8192 u64 keys (64 KB LDS,
// fine on 320 KB/WGP), then gather the sorted top-4096 boxes/scores/labels.
// ---------------------------------------------------------------------------
__global__ __launch_bounds__(1024) void k_sort_topk(
    const unsigned long long* __restrict__ cand_keys,
    const float* __restrict__ scores, const float4* __restrict__ boxes,
    float* __restrict__ topk_score, float4* __restrict__ topk_box,
    int* __restrict__ topk_label) {
  const int b = blockIdx.x, t = threadIdx.x;
  __shared__ unsigned long long sk[MSORT];
  for (int p = t; p < MSORT; p += 1024) sk[p] = cand_keys[b * MSORT + p];
  __syncthreads();
  for (int k = 2; k <= MSORT; k <<= 1) {
    for (int j = k >> 1; j > 0; j >>= 1) {
      for (int e = t; e < MSORT; e += 1024) {
        const int ixj = e ^ j;
        if (ixj > e) {
          const unsigned long long a = sk[e], c2 = sk[ixj];
          const bool up = ((e & k) == 0);
          if ((a > c2) == up) { sk[e] = c2; sk[ixj] = a; }
        }
      }
      __syncthreads();
    }
  }
  for (int p = t; p < KTOP; p += 1024) {
    const unsigned long long key = sk[p];
    float s; float4 bx; int lab;
    if (key == ~0ULL) {
      s = -1.0f; bx = make_float4(0.f, 0.f, 0.f, 0.f); lab = 0;
    } else {
      const unsigned idx = (unsigned)key;
      s = scores[b * NCAND + idx];
      bx = boxes[(size_t)b * NCAND + idx];
      lab = (int)(idx % 80u) + 1;
    }
    topk_score[b * KTOP + p] = s;
    topk_box[b * KTOP + p] = bx;
    topk_label[b * KTOP + p] = lab;
  }
}

// ---------------------------------------------------------------------------
// K5: NMS suppression bitmask. Block = 64 threads = 2 waves; each wave owns a
// 16(i) x 16(j) tile. The union denominator area_i + area_j for the tile is
// computed as a rank-2 outer product on the matrix pipe (V_WMMA). Intersection
// + threshold on VALU. Row bits ballot'ed through LDS atomicOr into one
// 32-bit word per row per block -> mask[B][4096][128].
// ---------------------------------------------------------------------------
__global__ __launch_bounds__(64) void k_nms_mask(
    const float4* __restrict__ topk_box, const int* __restrict__ topk_label,
    unsigned* __restrict__ maskWords) {
  const int b = blockIdx.z;
  const int i0 = blockIdx.y * 16;
  const int jb = blockIdx.x * 32;
  const int t = threadIdx.x;

  if (jb + 31 <= i0) {  // whole tile has j <= i -> no suppression bits
    if (t < 16)
      maskWords[((size_t)b * KTOP + i0 + t) * 128 + blockIdx.x] = 0u;
    return;
  }

  __shared__ float4 ibx[16];
  __shared__ int ilab[16];
  __shared__ float iarea[16];
  __shared__ unsigned rowbits[2][16];

  if (t < 16) {
    const float4 bx = topk_box[b * KTOP + i0 + t];
    ibx[t] = bx;
    ilab[t] = topk_label[b * KTOP + i0 + t];
    iarea[t] = (bx.z - bx.x) * (bx.w - bx.y);
  }
  if (t < 32) rowbits[t >> 4][t & 15] = 0u;
  __syncthreads();

  const int wv = t >> 5;          // which 16-wide j sub-tile this wave owns
  const int lane = t & 31;
  const int n2 = lane & 15;       // column within tile
  const int hi = lane >> 4;       // D rows r (hi=0) vs r+8 (hi=1)
  const int jg = jb + wv * 16 + n2;

  const float4 bj = topk_box[b * KTOP + jg];
  const int labj = topk_label[b * KTOP + jg];
  const float areaJ = (bj.z - bj.x) * (bj.w - bj.y);
  const float h0 = (hi == 0) ? 1.0f : 0.0f;  // lanes 16-31 carry zero K-cols

  v8f dsum;
  float sumScale;
#if WMMA_MODE == 2
  // A (16x4): lane m<16 holds {A[m][0],A[m][1]}={area_i,1}; lanes 16-31 K=2,3 = 0
  // B (4x16): lane n<16 holds {B[0][n],B[1][n]}={1,area_j};  lanes 16-31 K=2,3 = 0
  v2f av, bv2;
  av[0] = iarea[n2] * h0;  av[1] = h0;
  bv2[0] = h0;             bv2[1] = areaJ * h0;
  v8f cz = {};
  dsum = __builtin_amdgcn_wmma_f32_16x16x4_f32(false, av, false, bv2,
                                               (short)0, cz, false, false);
  sumScale = 1.0f;
#elif WMMA_MODE == 1
  // f16 path: areas pre-scaled by 1/64 to stay in f16 range (max ~16663)
  v16h a16 = {}, b16 = {};
  a16[0] = (_Float16)(iarea[n2] * 0.015625f * h0);
  a16[1] = (_Float16)h0;
  b16[0] = (_Float16)h0;
  b16[1] = (_Float16)(areaJ * 0.015625f * h0);
  v8f cz = {};
  dsum = __builtin_amdgcn_wmma_f32_16x16x32_f16(false, a16, false, b16,
                                                (short)0, cz, false, false);
  sumScale = 64.0f;
#else
  #pragma unroll
  for (int r = 0; r < 8; ++r) dsum[r] = iarea[r + 8 * hi] + areaJ;
  sumScale = 1.0f;
#endif

  #pragma unroll
  for (int r = 0; r < 8; ++r) {
    const int local = r + 8 * hi;     // D-matrix row held by this lane/reg
    const int ig = i0 + local;
    if ((jg > ig) && (ilab[local] == labj)) {
      const float4 bi = ibx[local];
      float iw = fminf(bi.z, bj.z) - fmaxf(bi.x, bj.x); iw = fmaxf(iw, 0.f);
      float ih = fminf(bi.w, bj.w) - fmaxf(bi.y, bj.y); ih = fmaxf(ih, 0.f);
      const float inter = iw * ih;
      const float denom = dsum[r] * sumScale - inter + 1e-9f;
      if (inter * 2.0f > denom)        // iou > 0.5
        atomicOr(&rowbits[wv][local], 1u << n2);
    }
  }
  __syncthreads();
  if (t < 16) {
    const unsigned wbits = rowbits[0][t] | (rowbits[1][t] << 16);
    maskWords[((size_t)b * KTOP + i0 + t) * 128 + blockIdx.x] = wbits;
  }
}

// ---------------------------------------------------------------------------
// K6: greedy scan, one wave per image (wave-synchronous, no barriers).
// Each lane owns 4 of the 128 suppression words; keep decision for row i is
// made by the owner lane and broadcast with __shfl.
// ---------------------------------------------------------------------------
__global__ __launch_bounds__(32) void k_nms_scan(
    const float* __restrict__ topk_score,
    const unsigned* __restrict__ maskWords,
    unsigned* __restrict__ keepWords) {
  const int b = blockIdx.x, lane = threadIdx.x;
  unsigned candw[4], removed[4] = {0u, 0u, 0u, 0u}, keepw[4] = {0u, 0u, 0u, 0u};
  #pragma unroll
  for (int slot = 0; slot < 4; ++slot) {
    const int w = lane + 32 * slot;
    unsigned cw = 0u;
    for (int bit = 0; bit < 32; ++bit)
      cw |= (topk_score[b * KTOP + w * 32 + bit] > SCORE_THRESH ? 1u : 0u) << bit;
    candw[slot] = cw;
  }
  for (int i = 0; i < KTOP; ++i) {
    const int w = i >> 5, bit = i & 31;
    const int owner = w & 31, slot = w >> 5;
    int keep = 0;
    if (lane == owner) {
      const unsigned cw = slot == 0 ? candw[0] : slot == 1 ? candw[1]
                        : slot == 2 ? candw[2] : candw[3];
      const unsigned rm = slot == 0 ? removed[0] : slot == 1 ? removed[1]
                        : slot == 2 ? removed[2] : removed[3];
      keep = (int)(((cw >> bit) & 1u) && !((rm >> bit) & 1u));
    }
    keep = __shfl(keep, owner, 32);
    if (keep) {
      const unsigned* row = maskWords + ((size_t)b * KTOP + i) * 128;
      removed[0] |= row[lane];
      removed[1] |= row[lane + 32];
      removed[2] |= row[lane + 64];
      removed[3] |= row[lane + 96];
      if (lane == owner) {
        const unsigned m = 1u << bit;
        if (slot == 0) keepw[0] |= m; else if (slot == 1) keepw[1] |= m;
        else if (slot == 2) keepw[2] |= m; else keepw[3] |= m;
      }
    }
  }
  #pragma unroll
  for (int slot = 0; slot < 4; ++slot)
    keepWords[b * 128 + lane + 32 * slot] = keepw[slot];
}

// ---------------------------------------------------------------------------
// K7: emit first 100 kept detections (already score-descending), zero-fill
// the rest. Output layout matches the flattened tuple (boxes, scores, labels).
// ---------------------------------------------------------------------------
__global__ __launch_bounds__(32) void k_finalize(
    const float* __restrict__ topk_score, const float4* __restrict__ topk_box,
    const int* __restrict__ topk_label, const unsigned* __restrict__ keepWords,
    float* __restrict__ out) {
  const int b = blockIdx.x;
  if (threadIdx.x != 0) return;
  float* out_boxes = out;                             // [B][100][4]
  float* out_scores = out + BIMG * NDET * 4;          // [B][100]
  float* out_labels = out + BIMG * NDET * 4 + BIMG * NDET;
  int cnt = 0;
  for (int i = 0; i < KTOP && cnt < NDET; ++i) {
    if ((keepWords[b * 128 + (i >> 5)] >> (i & 31)) & 1u) {
      const float4 bx = topk_box[b * KTOP + i];
      out_boxes[(b * NDET + cnt) * 4 + 0] = bx.x;
      out_boxes[(b * NDET + cnt) * 4 + 1] = bx.y;
      out_boxes[(b * NDET + cnt) * 4 + 2] = bx.z;
      out_boxes[(b * NDET + cnt) * 4 + 3] = bx.w;
      out_scores[b * NDET + cnt] = topk_score[b * KTOP + i];
      out_labels[b * NDET + cnt] = (float)topk_label[b * KTOP + i];
      ++cnt;
    }
  }
  for (; cnt < NDET; ++cnt) {
    out_boxes[(b * NDET + cnt) * 4 + 0] = 0.f;
    out_boxes[(b * NDET + cnt) * 4 + 1] = 0.f;
    out_boxes[(b * NDET + cnt) * 4 + 2] = 0.f;
    out_boxes[(b * NDET + cnt) * 4 + 3] = 0.f;
    out_scores[b * NDET + cnt] = 0.f;
    out_labels[b * NDET + cnt] = 0.f;
  }
}

// ---------------------------------------------------------------------------
extern "C" void kernel_launch(void* const* d_in, const int* in_sizes, int n_in,
                              void* d_out, int out_size, void* d_ws,
                              size_t ws_size, hipStream_t stream) {
  (void)in_sizes; (void)n_in; (void)out_size;
  const float* logits = (const float*)d_in[0];
  const float* regs   = (const float*)d_in[1];
  const float* rois   = (const float*)d_in[2];
  const int*   pH     = (const int*)d_in[3];
  const int*   pW     = (const int*)d_in[4];

  size_t off = 0;
  auto carve = [&](size_t bytes) -> char* {
    char* p = (char*)d_ws + off;
    off += (bytes + 255) & ~(size_t)255;
    return p;
  };
  float*  scores     = (float*) carve((size_t)BIMG * NCAND * 4);
  float4* boxes      = (float4*)carve((size_t)BIMG * NCAND * 16);
  float*  topk_score = (float*) carve((size_t)BIMG * KTOP * 4);
  float4* topk_box   = (float4*)carve((size_t)BIMG * KTOP * 16);
  int*    topk_label = (int*)   carve((size_t)BIMG * KTOP * 4);
  unsigned long long* cand_keys =
      (unsigned long long*)carve((size_t)BIMG * MSORT * 8);
  unsigned* cand_count = (unsigned*)carve(BIMG * 4);
  int*      selBin     = (int*)     carve(BIMG * 4);
  unsigned* keepWords  = (unsigned*)carve((size_t)BIMG * 128 * 4);
  unsigned* maskWords  = (unsigned*)carve((size_t)BIMG * KTOP * 128 * 4);
  if (off > ws_size) return;  // workspace too small; nothing safe to do

  k_score_decode<<<BIMG * NROI, 128, 0, stream>>>(logits, regs, rois, pH, pW,
                                                  scores, boxes);
  k_hist_select<<<BIMG, 1024, 0, stream>>>(scores, selBin, cand_count,
                                           cand_keys);
  k_compact<<<(BIMG * NCAND + 255) / 256, 256, 0, stream>>>(scores, selBin,
                                                            cand_count,
                                                            cand_keys);
  k_sort_topk<<<BIMG, 1024, 0, stream>>>(cand_keys, scores, boxes, topk_score,
                                         topk_box, topk_label);
  dim3 gmask(KTOP / 32, KTOP / 16, BIMG);  // 128 x 256 x 4 blocks of 64
  k_nms_mask<<<gmask, 64, 0, stream>>>(topk_box, topk_label, maskWords);
  k_nms_scan<<<BIMG, 32, 0, stream>>>(topk_score, maskWords, keepWords);
  k_finalize<<<BIMG, 32, 0, stream>>>(topk_score, topk_box, topk_label,
                                      keepWords, (float*)d_out);
}
